// GuidedDiffusionRGC_84387517432643
// MI455X (gfx1250) — matrix operand order repaired and checked by reading, not compiled, not run on hardware
//
#include <hip/hip_runtime.h>
#include <hip/hip_bf16.h>

// ---------------- problem constants ----------------
#define NN 32768      // B*N nodes
#define DD 256        // feature dim
#define EE 524288     // edges
#define RR 10         // relations
#define NBASE 4       // bases
#define KCOLS 1280    // (NBASE+1)*256 fused weight columns

typedef __attribute__((ext_vector_type(16))) __bf16 v16bf;
typedef __attribute__((ext_vector_type(8)))  float  v8f;

union Frag16 { v16bf v; uint4 q[2]; };
union Pack16 { __bf16 h[16]; uint4 q[2]; };

// ---------------- workspace layout (bytes) ----------------
// xb   : [NN][1024] f32   = 128 MiB   (basis-projected features, L2-resident)
// agg  : [NN][256]  f32   =  32 MiB   (layer-0 accumulator / hidden)
// cnt  : [RR*NN]    i32   = 1.25 MiB
// BwT  : [1280][256] bf16 = 640 KiB   (transposed fused weights)
#define XB_OFF   ((size_t)0)
#define AGG_OFF  ((size_t)134217728)
#define CNT_OFF  ((size_t)167772160)
#define BWT_OFF  ((size_t)169082880)

// ---------------- small utility kernels ----------------
__global__ void rgcn_zero_i32(int* p, int n) {
    int i = blockIdx.x * blockDim.x + threadIdx.x;
    if (i < n) p[i] = 0;
}

__global__ void rgcn_count_edges(const int* __restrict__ ec,
                                 const int* __restrict__ rel,
                                 int* __restrict__ cnt) {
    int e = blockIdx.x * blockDim.x + threadIdx.x;
    if (e < EE) {
        int d = ec[EE + e];
        int r = rel[e];
        atomicAdd(&cnt[r * NN + d], 1);
    }
}

__global__ void rgcn_relu_inplace(float* __restrict__ p, int n) {
    int i = blockIdx.x * blockDim.x + threadIdx.x;
    if (i < n) {
        float v = p[i];
        p[i] = v > 0.0f ? v : 0.0f;
    }
}

// Build transposed fused weights: BwT[col][k], col 0..1023 = basis[b][k][o]
// (col = b*256+o), col 1024..1279 = root[k][col-1024].  f32 -> bf16.
__global__ void rgcn_prep_weights(const float* __restrict__ basis,
                                  const float* __restrict__ root,
                                  __bf16* __restrict__ BwT) {
    int col = blockIdx.x;       // 0..1279
    int k   = threadIdx.x;      // 0..255
    float v;
    if (col < 1024) {
        int b = col >> 8, o = col & 255;
        v = basis[(size_t)b * 65536 + (size_t)k * 256 + o];
    } else {
        v = root[(size_t)k * 256 + (col - 1024)];
    }
    BwT[(size_t)col * 256 + k] = (__bf16)v;
}

// ---------------- fused WMMA GEMM ----------------
// C[N,1280] = A_f32[N,256] (->bf16) @ Wcat_bf16[256,1280]
// cols <1024 -> xb ;  cols >=1024 -> agg = bias + A@root  (full overwrite)
// block = 256 threads = 8 waves; block tile 128x64; wave tile 32x32 (2x2 wmma)
__global__ __launch_bounds__(256)
void rgcn_gemm_wmma(const float* __restrict__ A,
                    const __bf16* __restrict__ BwT,
                    float* __restrict__ xbOut,
                    float* __restrict__ aggOut,
                    const float* __restrict__ bias) {
    __shared__ __bf16 As[128][40];   // [row][k] pad 32->40 (80B row stride, 16B aligned)
    __shared__ __bf16 Bs[64][40];    // [col][k] (weights pre-transposed)

    const int t       = threadIdx.x;
    const int lane    = t & 31;
    const int wv      = t >> 5;
    const int wm      = (wv >> 1) * 32;    // wave row offset in tile: 0,32,64,96
    const int wn      = (wv & 1) * 32;     // wave col offset: 0,32
    const int half    = lane >> 4;
    const int l16     = lane & 15;
    const int rowBase = blockIdx.y * 128;
    const int colBase = blockIdx.x * 64;

    // A loader mapping: each thread loads 16 f32 (row ar, k chunk ac)
    const int ar = t >> 1;
    const int ac = (t & 1) * 16;
    // B loader mapping: each thread loads 8 bf16 (col bn, k chunk bk)
    const int bn = t >> 2;
    const int bk = (t & 3) * 8;

    v8f c00 = {}, c01 = {}, c10 = {}, c11 = {};

    for (int kb = 0; kb < 256; kb += 32) {
        // ---- stage A tile: f32 -> bf16 into LDS ----
        const float* ap = A + (size_t)(rowBase + ar) * 256 + kb + ac;
        float4 f0 = ((const float4*)ap)[0];
        float4 f1 = ((const float4*)ap)[1];
        float4 f2 = ((const float4*)ap)[2];
        float4 f3 = ((const float4*)ap)[3];
        if (kb < 224) __builtin_prefetch(ap + 32, 0, 1);   // global_prefetch_b8 next K tile
        Pack16 pk;
        pk.h[0]=(__bf16)f0.x;  pk.h[1]=(__bf16)f0.y;  pk.h[2]=(__bf16)f0.z;  pk.h[3]=(__bf16)f0.w;
        pk.h[4]=(__bf16)f1.x;  pk.h[5]=(__bf16)f1.y;  pk.h[6]=(__bf16)f1.z;  pk.h[7]=(__bf16)f1.w;
        pk.h[8]=(__bf16)f2.x;  pk.h[9]=(__bf16)f2.y;  pk.h[10]=(__bf16)f2.z; pk.h[11]=(__bf16)f2.w;
        pk.h[12]=(__bf16)f3.x; pk.h[13]=(__bf16)f3.y; pk.h[14]=(__bf16)f3.z; pk.h[15]=(__bf16)f3.w;
        *(uint4*)&As[ar][ac]     = pk.q[0];
        *(uint4*)&As[ar][ac + 8] = pk.q[1];

        // ---- stage B tile: bf16 direct b128 copy ----
        *(uint4*)&Bs[bn][bk] =
            *(const uint4*)(BwT + (size_t)(colBase + bn) * 256 + kb + bk);

        __syncthreads();

        // ---- fragments per ISA VGPR layouts ----
        Frag16 a0, a1, b0, b1;
        // A 16x32 bf16: lane row = M, K runs [8h,8h+8) and [16+8h,16+8h+8)
        a0.q[0] = *(const uint4*)&As[wm + l16][8 * half];
        a0.q[1] = *(const uint4*)&As[wm + l16][16 + 8 * half];
        a1.q[0] = *(const uint4*)&As[wm + 16 + l16][8 * half];
        a1.q[1] = *(const uint4*)&As[wm + 16 + l16][16 + 8 * half];
        // B 32x16 bf16: lane col = N, K runs [8h,8h+8) and [16+8h,16+8h+8)
        b0.q[0] = *(const uint4*)&Bs[wn + l16][8 * half];
        b0.q[1] = *(const uint4*)&Bs[wn + l16][16 + 8 * half];
        b1.q[0] = *(const uint4*)&Bs[wn + 16 + l16][8 * half];
        b1.q[1] = *(const uint4*)&Bs[wn + 16 + l16][16 + 8 * half];

        c00 = __builtin_amdgcn_wmma_f32_16x16x32_bf16(false, a0.v, false, b0.v, (short)0, c00, false, false);
        c01 = __builtin_amdgcn_wmma_f32_16x16x32_bf16(false, a0.v, false, b1.v, (short)0, c01, false, false);
        c10 = __builtin_amdgcn_wmma_f32_16x16x32_bf16(false, a1.v, false, b0.v, (short)0, c10, false, false);
        c11 = __builtin_amdgcn_wmma_f32_16x16x32_bf16(false, a1.v, false, b1.v, (short)0, c11, false, false);

        __syncthreads();
    }

    // ---- epilogue: C/D layout VGPR v -> M = v + 8*half, N = l16 ----
    const bool toAgg = (colBase >= 1024);
    auto storeTile = [&](const v8f& cc, int mt, int nt) {
        int col = colBase + wn + nt * 16 + l16;
        #pragma unroll
        for (int v = 0; v < 8; ++v) {
            int row = rowBase + wm + mt * 16 + v + 8 * half;
            float val = cc[v];
            if (!toAgg) {
                xbOut[(size_t)row * 1024 + col] = val;
            } else {
                int a = col - 1024;
                aggOut[(size_t)row * 256 + a] = bias[a] + val;
            }
        }
    };
    storeTile(c00, 0, 0);
    storeTile(c01, 0, 1);
    storeTile(c10, 1, 0);
    storeTile(c11, 1, 1);
}

// ---------------- per-edge message + scatter ----------------
// one wave per edge; xb gathers hit L2 (128 MiB < 192 MiB)
__global__ __launch_bounds__(256)
void rgcn_edge_scatter(const int* __restrict__ ec,
                       const int* __restrict__ rel,
                       const float* __restrict__ comp,
                       const int* __restrict__ cnt,
                       const float* __restrict__ xb,
                       float* __restrict__ agg) {
    const int lane = threadIdx.x & 31;
    const long e = (long)blockIdx.x * 8 + (threadIdx.x >> 5);
    if (e >= EE) return;
    const int s = ec[e];
    const int d = ec[EE + e];
    const int r = rel[e];
    const int c = cnt[r * NN + d];
    const float inv = (c > 0) ? (1.0f / (float)c) : 0.0f;
    const float w0 = comp[r * 4 + 0] * inv;
    const float w1 = comp[r * 4 + 1] * inv;
    const float w2 = comp[r * 4 + 2] * inv;
    const float w3 = comp[r * 4 + 3] * inv;

    const float4* xp = (const float4*)(xb + (size_t)s * 1024);
    float* ag = agg + (size_t)d * 256;

    #pragma unroll
    for (int rep = 0; rep < 2; ++rep) {
        int i = lane + rep * 32;            // float4 index within a 256-wide basis row
        float4 a = xp[i];                   // basis 0
        float4 b = xp[64 + i];              // basis 1
        float4 g = xp[128 + i];             // basis 2
        float4 h = xp[192 + i];             // basis 3
        float mx = w0 * a.x + w1 * b.x + w2 * g.x + w3 * h.x;
        float my = w0 * a.y + w1 * b.y + w2 * g.y + w3 * h.y;
        float mz = w0 * a.z + w1 * b.z + w2 * g.z + w3 * h.z;
        float mw = w0 * a.w + w1 * b.w + w2 * g.w + w3 * h.w;
        float* p = ag + i * 4;
        unsafeAtomicAdd(p + 0, mx);
        unsafeAtomicAdd(p + 1, my);
        unsafeAtomicAdd(p + 2, mz);
        unsafeAtomicAdd(p + 3, mw);
    }
}

// ---------------- driver ----------------
extern "C" void kernel_launch(void* const* d_in, const int* in_sizes, int n_in,
                              void* d_out, int out_size, void* d_ws, size_t ws_size,
                              hipStream_t stream) {
    const float* x      = (const float*)d_in[0];
    const int*   ec     = (const int*)d_in[3];
    const int*   rel    = (const int*)d_in[4];
    const float* basis0 = (const float*)d_in[5];
    const float* comp0  = (const float*)d_in[6];
    const float* root0  = (const float*)d_in[7];
    const float* bias0  = (const float*)d_in[8];
    const float* basis1 = (const float*)d_in[9];
    const float* comp1  = (const float*)d_in[10];
    const float* root1  = (const float*)d_in[11];
    const float* bias1  = (const float*)d_in[12];

    char* ws = (char*)d_ws;
    float*  xb   = (float*)(ws + XB_OFF);
    float*  agg0 = (float*)(ws + AGG_OFF);
    int*    cnt  = (int*)(ws + CNT_OFF);
    __bf16* BwT  = (__bf16*)(ws + BWT_OFF);
    float*  out  = (float*)d_out;

    // segment counts (same for both layers)
    rgcn_zero_i32<<<(RR * NN + 255) / 256, 256, 0, stream>>>(cnt, RR * NN);
    rgcn_count_edges<<<EE / 256, 256, 0, stream>>>(ec, rel, cnt);

    dim3 ggrid(KCOLS / 64, NN / 128);

    // ---- layer 0 ----
    rgcn_prep_weights<<<KCOLS, 256, 0, stream>>>(basis0, root0, BwT);
    rgcn_gemm_wmma<<<ggrid, 256, 0, stream>>>(x, BwT, xb, agg0, bias0);
    rgcn_edge_scatter<<<EE / 8, 256, 0, stream>>>(ec, rel, comp0, cnt, xb, agg0);
    rgcn_relu_inplace<<<(NN * DD) / 256, 256, 0, stream>>>(agg0, NN * DD);

    // ---- layer 1 (writes directly into d_out) ----
    rgcn_prep_weights<<<KCOLS, 256, 0, stream>>>(basis1, root1, BwT);
    rgcn_gemm_wmma<<<ggrid, 256, 0, stream>>>(agg0, BwT, xb, out, bias1);
    rgcn_edge_scatter<<<EE / 8, 256, 0, stream>>>(ec, rel, comp1, cnt, xb, out);
}
